// AttentionHead_Hybrid1_8315056685229
// MI455X (gfx1250) — matrix operand order repaired and checked by reading, not compiled, not run on hardware
//
#include <hip/hip_runtime.h>

// B=64, S=1024, D=64. Output: f32 [B,S,D].
// ws layout: W (64*64 f32, 16KB) | U f16 8MB | Km f16 8MB | V f16 8MB | rowscale f32 256KB

typedef __attribute__((ext_vector_type(16))) _Float16 v16h;
typedef __attribute__((ext_vector_type(8)))  float    v8f;

#define Bb 64
#define Ss 1024
#define Dd 64
#define LSTR 68   // LDS row stride in halves (padded: 136B rows, 8B aligned)

union AF { v16h v; unsigned u[8]; };
union CF { v8f  v; float    f[8]; };

// 16-bit A-matrix 16x32: VGPR v<4: K = 8*hi + 2v ; v>=4: K = 16 + 8*hi + 2(v-4)
__device__ __forceinline__ int a_kidx(int v, int hi) {
  return (v < 4) ? (hi * 8 + 2 * v) : (16 + hi * 8 + 2 * (v - 4));
}
// 16-bit B-matrix 32x16: VGPR v<4: K = 8*hi + 2v ; v>=4: K = 16 + 8*hi + 2(v-4) ; N = lane&15
__device__ __forceinline__ int b_kidx(int v, int hi) { return a_kidx(v, hi); }

// ---------------------------------------------------------------------------
// Kernel 1: build the Givens-circuit orthogonal W (sequential chain of 125
// adjacent-row rotations on identity). One wave; each lane owns 2 columns, so
// rotations (which mix ROWS i,i+1 column-wise) have no cross-lane deps.
// ---------------------------------------------------------------------------
__global__ void build_W_kernel(const float* __restrict__ phi, float* __restrict__ Wout) {
  __shared__ float W[Dd][Dd];
  const int lane = threadIdx.x;             // 0..31
  for (int r = 0; r < Dd; ++r) {
    W[r][lane]      = (r == lane)      ? 1.0f : 0.0f;
    W[r][lane + 32] = (r == lane + 32) ? 1.0f : 0.0f;
  }
  for (int k = 0; k < 2 * Dd - 3; ++k) {
    const int i = (k < Dd - 1) ? k : (2 * Dd - 4 - k);  // 0..62 then 61..0
    const float a = phi[k];
    const float c = __cosf(a), s = __sinf(a);
    float ri0 = W[i][lane],      rj0 = W[i + 1][lane];
    float ri1 = W[i][lane + 32], rj1 = W[i + 1][lane + 32];
    W[i][lane]          =  c * ri0 + s * rj0;
    W[i + 1][lane]      = -s * ri0 + c * rj0;
    W[i][lane + 32]     =  c * ri1 + s * rj1;
    W[i + 1][lane + 32] = -s * ri1 + c * rj1;
  }
  for (int r = 0; r < Dd; ++r) {
    Wout[r * Dd + lane]      = W[r][lane];
    Wout[r * Dd + lane + 32] = W[r][lane + 32];
  }
}

// ---------------------------------------------------------------------------
// Kernel 2: per-token prep. 8 waves/block, 1 token/wave.
//   U[j]  = x_j / n_j                      (f16 queries)
//   Km[j] = sqrt(n_j) * (W^T u_j)          (f16 keys; folds sqrt(n_j))
//   V[j]  = x_j Wv^T + bv                  (f16 values)
//   rs[j] = n_j / 8                        (row scale incl. 1/sqrt(D))
// ---------------------------------------------------------------------------
__global__ void __launch_bounds__(256)
prep_kernel(const float* __restrict__ x, const float* __restrict__ Wv,
            const float* __restrict__ bv, const float* __restrict__ W,
            _Float16* __restrict__ U, _Float16* __restrict__ Km,
            _Float16* __restrict__ Vm, float* __restrict__ rs) {
  __shared__ float Wl[Dd * Dd];
  __shared__ float Wvl[Dd * Dd];
  __shared__ float bvl[Dd];
  __shared__ float ub[8][Dd];
  const int t = threadIdx.x;
  for (int i = t; i < Dd * Dd; i += 256) { Wl[i] = W[i]; Wvl[i] = Wv[i]; }
  if (t < Dd) bvl[t] = bv[t];
  __syncthreads();

  const int wave = t >> 5, lane = t & 31;
  const long tok = (long)blockIdx.x * 8 + wave;
  const float* xr = x + tok * Dd;
  const float x0 = xr[lane], x1 = xr[lane + 32];
  float ss = x0 * x0 + x1 * x1;
  for (int m = 16; m >= 1; m >>= 1) ss += __shfl_xor(ss, m, 32);
  const float n   = sqrtf(ss);
  const float inv = 1.0f / fmaxf(n, 1e-12f);
  const float u0 = x0 * inv, u1 = x1 * inv;
  ub[wave][lane] = u0; ub[wave][lane + 32] = u1;
  __syncthreads();

  _Float16* Ur = U + tok * Dd;
  Ur[lane] = (_Float16)u0; Ur[lane + 32] = (_Float16)u1;
  const float sq = sqrtf(n);
  for (int half = 0; half < 2; ++half) {
    const int e = lane + half * 32;
    float y = 0.0f, v = 0.0f;
    for (int d = 0; d < Dd; ++d) {
      const float ud = ub[wave][d];
      y = fmaf(ud, Wl[d * Dd + e], y);       // (W^T u)_e
      v = fmaf(ud, Wvl[e * Dd + d], v);      // (u Wv^T)_e
    }
    Km[tok * Dd + e] = (_Float16)(sq * y);
    Vm[tok * Dd + e] = (_Float16)(fmaf(n, v, bvl[e]));
  }
  if (lane == 0) rs[tok] = n * 0.125f;
}

// ---------------------------------------------------------------------------
// Kernel 3: fused flash attention + residual + LayerNorm.
// Block = 128 threads (4 waves); wave w owns query rows [blk*64 + 16w, +16).
// Double-buffered LDS pipeline over 16 column tiles of 64: stage tile jt+1
// while computing tile jt (one barrier per iteration), prefetch tile jt+2.
// ---------------------------------------------------------------------------
__global__ void __launch_bounds__(128)
attn_kernel(const float* __restrict__ x, const _Float16* __restrict__ U,
            const _Float16* __restrict__ Km, const _Float16* __restrict__ Vm,
            const float* __restrict__ rs, float* __restrict__ out) {
  __shared__ _Float16 Kt[2][Dd * LSTR];      // [j][d] row-major (double buffer)
  __shared__ _Float16 Vt[2][Dd * LSTR];      // [e][j] transposed (double buffer)
  __shared__ _Float16 Pl[4][16 * LSTR];      // per-wave P transpose scratch

  const int t = threadIdx.x;
  const int wave = t >> 5, lane = t & 31;
  const int hi = (lane >= 16) ? 1 : 0;
  const int ln = lane & 15;
  const int b = blockIdx.y;
  const long rowbase = (long)b * Ss + blockIdx.x * 64 + wave * 16;

  // global -> LDS tile staging for column tile jt into buffer buf
  auto stage = [&](int jt, int buf) {
    const long cb = (long)b * Ss + jt * 64;
    {  // Kt: each thread copies 32 halves (8B chunks) of one half-row
      const int r = t >> 1, c = (t & 1) * 32;
      const unsigned long long* src = (const unsigned long long*)(Km + (cb + r) * Dd + c);
      for (int q = 0; q < 8; ++q)
        *(unsigned long long*)(&Kt[buf][r * LSTR + c + q * 4]) = src[q];
    }
    {  // Vt transposed: lane-coalesced global reads, scatter to [e][j]
      const int e = t & 63, j0 = (t >> 6) * 32;
      for (int q = 0; q < 32; ++q)
        Vt[buf][e * LSTR + j0 + q] = Vm[(cb + j0 + q) * Dd + e];
    }
  };

  // Q A-fragments (16x32 halves each; K = d in [0,32) and [32,64))
  AF aq0, aq1;
  {
    const _Float16* qr = U + (rowbase + ln) * Dd;
    for (int v = 0; v < 8; ++v) {
      aq0.u[v] = *(const unsigned*)(qr + a_kidx(v, hi));
      aq1.u[v] = *(const unsigned*)(qr + 32 + a_kidx(v, hi));
    }
  }
  float rs8[8];
  for (int v = 0; v < 8; ++v) rs8[v] = rs[rowbase + v + 8 * hi];

  CF o[4];
  float mrow[8], lrow[8];
  for (int nn = 0; nn < 4; ++nn)
    for (int v = 0; v < 8; ++v) o[nn].f[v] = 0.0f;
  for (int v = 0; v < 8; ++v) { mrow[v] = -1e30f; lrow[v] = 0.0f; }

  stage(0, 0);
  __syncthreads();

  for (int jt = 0; jt < Ss / 64; ++jt) {
    const int buf = jt & 1;

    // kick off staging of the next tile into the other buffer (its previous
    // readers finished at the barrier ending iteration jt-1)
    if (jt + 1 < Ss / 64) stage(jt + 1, buf ^ 1);
    if (jt + 2 < Ss / 64) {  // pull tile jt+2 toward the WGP (global_prefetch_b8)
      const long pb = (long)b * Ss + (jt + 2) * 64 + (t >> 1);
      __builtin_prefetch(Km + pb * Dd, 0, 2);
      __builtin_prefetch(Vm + pb * Dd, 0, 2);
    }

    // ---- S = rs_i * (Q K^T)^2 for four 16x16 subtiles ----
    CF sc[4];
    for (int jj = 0; jj < 4; ++jj) {
      AF b0, b1;
      const _Float16* kr = &Kt[buf][(jj * 16 + ln) * LSTR];
      for (int v = 0; v < 8; ++v) {
        const int k0 = b_kidx(v, hi);
        b0.u[v] = *(const unsigned*)(kr + k0);
        b1.u[v] = *(const unsigned*)(kr + 32 + k0);
      }
      CF acc;
      for (int v = 0; v < 8; ++v) acc.f[v] = 0.0f;
      acc.v = __builtin_amdgcn_wmma_f32_16x16x32_f16(false, aq0.v, false, b0.v,
                                                     (short)0, acc.v, false, false);
      acc.v = __builtin_amdgcn_wmma_f32_16x16x32_f16(false, aq1.v, false, b1.v,
                                                     (short)0, acc.v, false, false);
      for (int v = 0; v < 8; ++v) {
        const float s = acc.f[v];
        sc[jj].f[v] = rs8[v] * s * s;
      }
    }

    // ---- online softmax update (row reductions within 16-lane halves) ----
    float lpart[8];
    for (int v = 0; v < 8; ++v) {
      float tm = fmaxf(fmaxf(sc[0].f[v], sc[1].f[v]), fmaxf(sc[2].f[v], sc[3].f[v]));
      for (int m = 8; m >= 1; m >>= 1) tm = fmaxf(tm, __shfl_xor(tm, m, 32));
      const float mn = fmaxf(mrow[v], tm);
      const float alpha = __expf(mrow[v] - mn);
      mrow[v] = mn;
      lrow[v] *= alpha;
      for (int nn = 0; nn < 4; ++nn) o[nn].f[v] *= alpha;
      float lp = 0.0f;
      for (int jj = 0; jj < 4; ++jj) {
        const float p = __expf(sc[jj].f[v] - mn);
        sc[jj].f[v] = p;
        lp += p;
      }
      lpart[v] = lp;
    }
    for (int v = 0; v < 8; ++v) {
      float lp = lpart[v];
      for (int m = 8; m >= 1; m >>= 1) lp += __shfl_xor(lp, m, 32);
      lrow[v] += lp;
    }

    // ---- C-layout -> A-layout transpose of P through per-wave LDS ----
    _Float16* pw = Pl[wave];
    for (int jj = 0; jj < 4; ++jj)
      for (int v = 0; v < 8; ++v)
        pw[(v + 8 * hi) * LSTR + jj * 16 + ln] = (_Float16)sc[jj].f[v];
    AF ap0, ap1;
    {
      const _Float16* pr = pw + ln * LSTR;
      for (int v = 0; v < 8; ++v) {
        ap0.u[v] = *(const unsigned*)(pr + a_kidx(v, hi));
        ap1.u[v] = *(const unsigned*)(pr + 32 + a_kidx(v, hi));
      }
    }

    // ---- O += P V ----
    for (int nn = 0; nn < 4; ++nn) {
      AF bv0, bv1;
      const _Float16* vr = &Vt[buf][(nn * 16 + ln) * LSTR];
      for (int v = 0; v < 8; ++v) {
        const int k0 = b_kidx(v, hi);
        bv0.u[v] = *(const unsigned*)(vr + k0);
        bv1.u[v] = *(const unsigned*)(vr + 32 + k0);
      }
      o[nn].v = __builtin_amdgcn_wmma_f32_16x16x32_f16(false, ap0.v, false, bv0.v,
                                                       (short)0, o[nn].v, false, false);
      o[nn].v = __builtin_amdgcn_wmma_f32_16x16x32_f16(false, ap1.v, false, bv1.v,
                                                       (short)0, o[nn].v, false, false);
    }

    // one barrier per iteration: staging of jt+1 done, readers of buf done
    __syncthreads();
  }

  // ---- epilogue: 1/l, residual, LayerNorm over D=64 ----
  float h[4][8], sh[8], sh2[8];
  for (int v = 0; v < 8; ++v) { sh[v] = 0.0f; sh2[v] = 0.0f; }
  for (int v = 0; v < 8; ++v) {
    const float invl = 1.0f / lrow[v];
    const long row = rowbase + v + 8 * hi;
    for (int nn = 0; nn < 4; ++nn) {
      const float a  = o[nn].f[v] * invl;
      const float hv = a + x[row * Dd + nn * 16 + ln];
      h[nn][v] = hv;
      sh[v]  += hv;
      sh2[v] += hv * hv;
    }
  }
  for (int v = 0; v < 8; ++v) {
    float s1 = sh[v], s2 = sh2[v];
    for (int m = 8; m >= 1; m >>= 1) {
      s1 += __shfl_xor(s1, m, 32);
      s2 += __shfl_xor(s2, m, 32);
    }
    const float mu  = s1 * (1.0f / 64.0f);
    const float var = s2 * (1.0f / 64.0f) - mu * mu;
    const float inv = rsqrtf(var + 1e-5f);
    const long row = rowbase + v + 8 * hi;
    for (int nn = 0; nn < 4; ++nn)
      out[row * Dd + nn * 16 + ln] = (h[nn][v] - mu) * inv;
  }
}

// ---------------------------------------------------------------------------
extern "C" void kernel_launch(void* const* d_in, const int* in_sizes, int n_in,
                              void* d_out, int out_size, void* d_ws, size_t ws_size,
                              hipStream_t stream) {
  (void)in_sizes; (void)n_in; (void)out_size; (void)ws_size;
  const float* x   = (const float*)d_in[0];
  const float* Wv  = (const float*)d_in[1];
  const float* bv  = (const float*)d_in[2];
  const float* phi = (const float*)d_in[3];

  char* ws = (char*)d_ws;
  const size_t half_mat = (size_t)Bb * Ss * Dd * sizeof(_Float16);  // 8 MB
  float*    Wbuf = (float*)ws;
  _Float16* U    = (_Float16*)(ws + 16384);
  _Float16* Km   = (_Float16*)(ws + 16384 + half_mat);
  _Float16* Vm   = (_Float16*)(ws + 16384 + 2 * half_mat);
  float*    rsc  = (float*)   (ws + 16384 + 3 * half_mat);

  build_W_kernel<<<1, 32, 0, stream>>>(phi, Wbuf);
  prep_kernel<<<(Bb * Ss) / 8, 256, 0, stream>>>(x, Wv, bv, Wbuf, U, Km, Vm, rsc);
  dim3 grid(Ss / 64, Bb);
  attn_kernel<<<grid, 128, 0, stream>>>(x, U, Km, Vm, rsc, (float*)d_out);
}
